// HetAgg_36438502539521
// MI455X (gfx1250) — compile-verified, matrix operand order checked by reading, not run on hardware
//
#include <hip/hip_runtime.h>
#include <hip/hip_bf16.h>
#include <math.h>

typedef __attribute__((ext_vector_type(16))) __bf16 v16bf;
typedef __attribute__((ext_vector_type(2)))  __bf16 v2bf;
typedef __attribute__((ext_vector_type(8)))  float  v8f;
typedef int vsi4 __attribute__((vector_size(16)));   // matches builtin param type

constexpr int D  = 128, T_ = 10, B_ = 4096;
constexpr int ND_ = 20000, FD_ = 2048;
constexpr int NG_ = 20000, FG_ = 1024;
constexpr int NCC_ = 10000, FC_ = 512;

#if defined(__has_builtin)
#if __has_builtin(__builtin_amdgcn_global_load_async_to_lds_b128)
#define HAVE_ASYNC_LDS 1
#endif
#endif

#define GLOBAL_AS __attribute__((address_space(1)))
#define LDS_AS    __attribute__((address_space(3)))

// 16-bit A/B WMMA fragment K-index for (lane, element i) per CDNA5 ISA 7.12.2.
// Note KMAP(ln, i+1) == KMAP(ln, i) + 1 for even i (pairs stay contiguous).
#define KMAP(ln, i) ( ((i) & 7) | (((i) & 8) << 1) | ((((ln)) & 16) >> 1) )

// ---------------------------------------------------------------------------
// prefrag_kernel: swizzle W[F,128] f32 -> global bf16 B-fragment buffer
// layout [F/32][8 ntile][32 lane][16], so proj blocks load operands b128.
// ---------------------------------------------------------------------------
__global__ void __launch_bounds__(256)
prefrag_kernel(const float* __restrict__ W, __bf16* __restrict__ Wf) {
  int idx = blockIdx.x * 256 + threadIdx.x;      // one bf16 half per thread
  int i  = idx & 15;
  int ln = (idx >> 4) & 31;
  int nt = (idx >> 9) & 7;
  int kt = idx >> 12;
  int k  = kt * 32 + KMAP(ln, i);
  int n  = nt * 16 + (ln & 15);
  Wf[idx] = (__bf16)W[(size_t)k * D + n];
}

// ---------------------------------------------------------------------------
// proj_kernel: out[N,128] = X[N,F] @ W + b. No LDS: A fragments are two runs
// of 8 consecutive k per lane -> 4x float4 global loads; B fragments come
// pre-swizzled from L2. grid.x = N/16, block = 128 (wave owns tiles {w,w+4}).
// ---------------------------------------------------------------------------
__global__ void __launch_bounds__(128)
proj_kernel(const float* __restrict__ X, const __bf16* __restrict__ Wf,
            const float* __restrict__ bias, float* __restrict__ out,
            int N, int F) {
  const int tid  = threadIdx.x;
  const int lane = tid & 31;
  const int wave = tid >> 5;
  const int row0 = blockIdx.x * 16;
  const int nc16 = lane & 15;
  const int hi   = lane >> 4;
  const float* xrow = X + (size_t)(row0 + nc16) * F + hi * 8;

  v8f acc0 = {}; v8f acc1 = {};
  for (int k0 = 0; k0 < F; k0 += 32) {
    __builtin_prefetch((const void*)(xrow + k0 + 128), 0, 1); // -> global_prefetch_b8
    const float4 xa = *(const float4*)(xrow + k0);
    const float4 xb = *(const float4*)(xrow + k0 + 4);
    const float4 xc = *(const float4*)(xrow + k0 + 16);
    const float4 xd = *(const float4*)(xrow + k0 + 20);
    v16bf a;
    a[0] = (__bf16)xa.x; a[1] = (__bf16)xa.y; a[2]  = (__bf16)xa.z; a[3]  = (__bf16)xa.w;
    a[4] = (__bf16)xb.x; a[5] = (__bf16)xb.y; a[6]  = (__bf16)xb.z; a[7]  = (__bf16)xb.w;
    a[8] = (__bf16)xc.x; a[9] = (__bf16)xc.y; a[10] = (__bf16)xc.z; a[11] = (__bf16)xc.w;
    a[12] = (__bf16)xd.x; a[13] = (__bf16)xd.y; a[14] = (__bf16)xd.z; a[15] = (__bf16)xd.w;
    const int kt = k0 >> 5;
    v16bf b0 = *(const v16bf*)(Wf + ((size_t)(kt * 8 + wave)     * 32 + lane) * 16);
    v16bf b1 = *(const v16bf*)(Wf + ((size_t)(kt * 8 + wave + 4) * 32 + lane) * 16);
    acc0 = __builtin_amdgcn_wmma_f32_16x16x32_bf16(false, a, false, b0, (short)0, acc0, false, false);
    acc1 = __builtin_amdgcn_wmma_f32_16x16x32_bf16(false, a, false, b1, (short)0, acc1, false, false);
  }
  const int n0 = wave * 16, n1 = (wave + 4) * 16;
  for (int j = 0; j < 8; ++j) {
    int m = row0 + j + 8 * hi;
    out[(size_t)m * D + n0 + nc16] = acc0[j] + bias[n0 + nc16];
    out[(size_t)m * D + n1 + nc16] = acc1[j] + bias[n1 + nc16];
  }
}

// ---------------------------------------------------------------------------
__global__ void gather_rows(const float* __restrict__ src, const int* __restrict__ ids,
                            float* __restrict__ dst) {
  int b = blockIdx.x;
  dst[(size_t)b * D + threadIdx.x] = src[(size_t)ids[b] * D + threadIdx.x];
}

// ---------------------------------------------------------------------------
// gru_cand_kernel: one (layer,type); block = 16 batch rows, 128 threads.
// W/U pre-swizzled to bf16 B-fragments in LDS (320KB/WGP budget), x staged
// via async global->LDS then packed into A-fragments; h kept in LDS as bf16
// already in A-fragment layout so each step reloads it with 8 ds_load_b128.
// ---------------------------------------------------------------------------
constexpr int FRAG_HALVES = 4 * 24 * 32 * 16;          // 49152 per matrix
constexpr size_t GRU_LDS_BYTES =
    (size_t)(2 * FRAG_HALVES + T_ * 4 * 32 * 16 + 2048) * 2  // W,U,X frags + h (bf16)
    + 2048 * 4                                                // xs scratch f32
    + (16 + 16 * T_) * 4                                      // cnt, mask
    + 16 * 4;                                                 // neighbor ids

__global__ void __launch_bounds__(128)
gru_cand_kernel(const float* __restrict__ proj, const int* __restrict__ neigh,
                const float* __restrict__ Wg, const float* __restrict__ Ug,
                const float* __restrict__ bg, float* __restrict__ cand) {
  extern __shared__ char smem[];
  __bf16* Wfrag = (__bf16*)smem;                     // [4][24][32][16]
  __bf16* Ufrag = Wfrag + FRAG_HALVES;
  __bf16* Xfrag = Ufrag + FRAG_HALVES;               // [T][4][32][16]
  __bf16* hbf   = Xfrag + T_ * 4 * 32 * 16;          // [4][32][16] A-frag layout
  float*  xs     = (float*)(hbf + 2048);             // [16][128]
  float*  cnt_l  = xs + 2048;                        // [16]
  float*  mask_l = cnt_l + 16;                       // [16][T]
  int*    nid_l  = (int*)(mask_l + 16 * T_);         // [16]

  const int tid  = threadIdx.x;
  const int lane = tid & 31;
  const int wave = tid >> 5;
  const int row0 = blockIdx.x * 16;
  const int nc16 = lane & 15;
  const int hi   = lane >> 4;

  // --- swizzle W,U into B-fragment layout (paired: cvt_pk + ds_store_b32) ---
  for (int idx2 = tid; idx2 < FRAG_HALVES / 2; idx2 += 128) {
    int i  = (idx2 & 7) * 2;
    int ln = (idx2 >> 3) & 31;
    int r  = idx2 >> 8;
    int nt = r % 24;
    int kf = r / 24;
    int k  = kf * 32 + KMAP(ln, i);
    int n  = nt * 16 + (ln & 15);
    size_t off = ((size_t)(kf * 24 + nt) * 32 + ln) * 16 + i;
    v2bf w2; w2[0] = (__bf16)Wg[k * 384 + n]; w2[1] = (__bf16)Wg[(k + 1) * 384 + n];
    v2bf u2; u2[0] = (__bf16)Ug[k * 384 + n]; u2[1] = (__bf16)Ug[(k + 1) * 384 + n];
    *(v2bf*)(Wfrag + off) = w2;
    *(v2bf*)(Ufrag + off) = u2;
  }
  // --- gather neighbor features (async global->LDS), build x A-fragments ---
  for (int t = 0; t < T_; ++t) {
    __syncthreads();
    if (tid < 16) nid_l[tid] = neigh[(row0 + tid) * T_ + t];
    __syncthreads();
    if (tid < 16) mask_l[tid * T_ + t] = (nid_l[tid] >= 0) ? 1.0f : 0.0f;
    for (int idx = tid; idx < 512; idx += 128) {      // 16B chunks of [16][128]f32
      int m = idx >> 5, c = idx & 31;
      int nid = nid_l[m];
#ifdef HAVE_ASYNC_LDS
      if (nid >= 0) {
        __builtin_amdgcn_global_load_async_to_lds_b128(
            (GLOBAL_AS vsi4*)(proj + (size_t)nid * D + c * 4),
            (LDS_AS vsi4*)(xs + idx * 4), 0, 0);
      } else {
        float4 z = {0.0f, 0.0f, 0.0f, 0.0f};
        *(float4*)(xs + idx * 4) = z;
      }
#else
      float4 v = {0.0f, 0.0f, 0.0f, 0.0f};
      if (nid >= 0) v = *(const float4*)(proj + (size_t)nid * D + c * 4);
      *(float4*)(xs + idx * 4) = v;
#endif
    }
#ifdef HAVE_ASYNC_LDS
    asm volatile("s_wait_asynccnt 0" ::: "memory");
#endif
    __syncthreads();
    for (int idx2 = tid; idx2 < 1024; idx2 += 128) {  // paired bf16 packing
      int i  = (idx2 & 7) * 2;
      int ln = (idx2 >> 3) & 31;
      int kf = idx2 >> 8;
      int k  = kf * 32 + KMAP(ln, i);
      const float2 v = *(const float2*)(xs + (ln & 15) * 128 + k);
      v2bf pv; pv[0] = (__bf16)v.x; pv[1] = (__bf16)v.y;
      *(v2bf*)(Xfrag + t * 2048 + kf * 512 + ln * 16 + i) = pv;
    }
  }
  __syncthreads();
  for (int i2 = tid; i2 < 1024; i2 += 128) ((unsigned*)hbf)[i2] = 0u;   // h0 = 0
  if (tid < 16) {
    float c = 0.0f;
    for (int t = 0; t < T_; ++t) c += mask_l[tid * T_ + t];
    cnt_l[tid] = fmaxf(c, 1.0f);
  }
  float bval[6];
  for (int s = 0; s < 6; ++s) bval[s] = bg[(wave + 4 * s) * 16 + nc16];
  // precompute inverse A-frag slot for this thread's two h column tiles
  int kfp[2], iip[2], hiap[2];
  for (int p = 0; p < 2; ++p) {
    int k = (wave + 4 * p) * 16 + nc16;
    kfp[p] = k >> 5;
    int kk = k & 31;
    hiap[p] = (kk >> 3) & 1;
    iip[p]  = (kk & 7) | ((kk & 16) >> 1);
  }

  v8f hold0 = {}, hold1 = {}, pool0 = {}, pool1 = {};
  const v8f vzero = {};

  for (int t = 0; t < T_; ++t) {
    __syncthreads();                                  // hbf up to date
    v16bf Ah[4];
    for (int kf = 0; kf < 4; ++kf)
      Ah[kf] = *(const v16bf*)(hbf + kf * 512 + lane * 16);

    v8f g[4], xn[2], hn[2];                           // z0,z1,r0,r1 | n gates
    for (int s = 0; s < 4; ++s) { v8f a; for (int j = 0; j < 8; ++j) a[j] = bval[s]; g[s] = a; }
    for (int p = 0; p < 2; ++p) { v8f a; for (int j = 0; j < 8; ++j) a[j] = bval[4 + p]; xn[p] = a; hn[p] = vzero; }

    for (int kf = 0; kf < 4; ++kf) {
      v16bf Ax = *(const v16bf*)(Xfrag + (t * 4 + kf) * 512 + lane * 16);
      for (int s = 0; s < 4; ++s) {
        int nt = wave + 4 * s;
        v16bf bw = *(const v16bf*)(Wfrag + ((kf * 24 + nt) * 32 + lane) * 16);
        v16bf bu = *(const v16bf*)(Ufrag + ((kf * 24 + nt) * 32 + lane) * 16);
        g[s] = __builtin_amdgcn_wmma_f32_16x16x32_bf16(false, Ax,     false, bw, (short)0, g[s], false, false);
        g[s] = __builtin_amdgcn_wmma_f32_16x16x32_bf16(false, Ah[kf], false, bu, (short)0, g[s], false, false);
      }
      for (int p = 0; p < 2; ++p) {
        int nt = wave + 4 * (4 + p);
        v16bf bw = *(const v16bf*)(Wfrag + ((kf * 24 + nt) * 32 + lane) * 16);
        v16bf bu = *(const v16bf*)(Ufrag + ((kf * 24 + nt) * 32 + lane) * 16);
        xn[p] = __builtin_amdgcn_wmma_f32_16x16x32_bf16(false, Ax,     false, bw, (short)0, xn[p], false, false);
        hn[p] = __builtin_amdgcn_wmma_f32_16x16x32_bf16(false, Ah[kf], false, bu, (short)0, hn[p], false, false);
      }
    }
    __syncthreads();                                  // all hbf reads done
    for (int p = 0; p < 2; ++p) {
      v8f& hold = p ? hold1 : hold0;
      v8f& pool = p ? pool1 : pool0;
      for (int j = 0; j < 8; ++j) {
        float z  = 1.0f / (1.0f + __expf(-g[p][j]));
        float r  = 1.0f / (1.0f + __expf(-g[2 + p][j]));
        float n  = tanhf(xn[p][j] + r * hn[p][j]);
        float hv = (1.0f - z) * n + z * hold[j];
        hold[j]  = hv;
        pool[j] += mask_l[(j + 8 * hi) * T_ + t] * hv;
        // store h back in A-fragment layout (inverse KMAP)
        hbf[kfp[p] * 512 + (((j + 8 * hi) | (hiap[p] << 4)) * 16) + iip[p]] = (__bf16)hv;
      }
    }
  }
  __syncthreads();
  for (int p = 0; p < 2; ++p) {
    const v8f& pool = p ? pool1 : pool0;
    for (int j = 0; j < 8; ++j) {
      int m = j + 8 * hi;
      cand[(size_t)(row0 + m) * D + (wave + 4 * p) * 16 + nc16] = pool[j] / cnt_l[m];
    }
  }
}

// ---------------------------------------------------------------------------
// attn_kernel: one wave32 per row. scores = leakyrelu([h|C_k]@att), softmax.
// ---------------------------------------------------------------------------
__global__ void __launch_bounds__(128)
attn_kernel(const float* __restrict__ h_in, const float* __restrict__ cand,
            const float* __restrict__ att, float* __restrict__ h_out) {
  const int wave = threadIdx.x >> 5, lane = threadIdx.x & 31;
  const int b = blockIdx.x * 4 + wave;
  const float4 hv = *(const float4*)(h_in + (size_t)b * D + lane * 4);
  const float4 a0 = *(const float4*)(att + lane * 4);
  const float4 a1 = *(const float4*)(att + 128 + lane * 4);
  float4 C[4];
  C[0] = hv;
  for (int t = 0; t < 3; ++t)
    C[1 + t] = *(const float4*)(cand + ((size_t)t * B_ + b) * D + lane * 4);
  float hdot = hv.x * a0.x + hv.y * a0.y + hv.z * a0.z + hv.w * a0.w;
  float s[4];
  for (int k = 0; k < 4; ++k)
    s[k] = C[k].x * a1.x + C[k].y * a1.y + C[k].z * a1.z + C[k].w * a1.w;
  for (int off = 16; off; off >>= 1) {
    hdot += __shfl_xor(hdot, off, 32);
    for (int k = 0; k < 4; ++k) s[k] += __shfl_xor(s[k], off, 32);
  }
  float mx = -1e30f;
  for (int k = 0; k < 4; ++k) {
    float v = hdot + s[k];
    v = (v >= 0.0f) ? v : 0.01f * v;                  // leaky_relu(0.01)
    s[k] = v; mx = fmaxf(mx, v);
  }
  float den = 0.0f;
  for (int k = 0; k < 4; ++k) { s[k] = __expf(s[k] - mx); den += s[k]; }
  float4 o = {0.0f, 0.0f, 0.0f, 0.0f};
  for (int k = 0; k < 4; ++k) {
    float w = s[k] / den;
    o.x += w * C[k].x; o.y += w * C[k].y; o.z += w * C[k].z; o.w += w * C[k].w;
  }
  *(float4*)(h_out + (size_t)b * D + lane * 4) = o;
}

// ---------------------------------------------------------------------------
extern "C" void kernel_launch(void* const* d_in, const int* in_sizes, int n_in,
                              void* d_out, int out_size, void* d_ws, size_t ws_size,
                              hipStream_t stream) {
  (void)in_sizes; (void)n_in; (void)out_size; (void)ws_size;
  const float* drug = (const float*)d_in[0];
  const float* gene = (const float*)d_in[1];
  const float* cell = (const float*)d_in[2];
  const float* Wd   = (const float*)d_in[3];
  const float* bd   = (const float*)d_in[4];
  const float* Wg   = (const float*)d_in[5];
  const float* bgn  = (const float*)d_in[6];
  const float* Wc   = (const float*)d_in[7];
  const float* bc   = (const float*)d_in[8];
  const float* gruW = (const float*)d_in[9];
  const float* gruU = (const float*)d_in[10];
  const float* grub = (const float*)d_in[11];
  const float* att  = (const float*)d_in[12];
  const int* ids    = (const int*)d_in[13];
  const int* neighD = (const int*)d_in[14];
  const int* neighG = (const int*)d_in[15];
  const int* neighC = (const int*)d_in[16];
  float* out = (float*)d_out;

  float* ws    = (float*)d_ws;
  float* projD = ws;                          // 20000*128
  float* projG = projD + (size_t)ND_ * D;     // 20000*128
  float* projC = projG + (size_t)NG_ * D;     // 10000*128
  float* hbuf  = projC + (size_t)NCC_ * D;    // 4096*128
  float* cand  = hbuf + (size_t)B_ * D;       // 6 * 4096*128
  __bf16* WfD  = (__bf16*)(cand + (size_t)6 * B_ * D);
  __bf16* WfG  = WfD + (size_t)FD_ * D;
  __bf16* WfC  = WfG + (size_t)FG_ * D;

  // 0) pre-swizzle projection weights into WMMA B-fragment layout (L2-resident)
  prefrag_kernel<<<FD_ * D / 256, 256, 0, stream>>>(Wd, WfD);
  prefrag_kernel<<<FG_ * D / 256, 256, 0, stream>>>(Wg, WfG);
  prefrag_kernel<<<FC_ * D / 256, 256, 0, stream>>>(Wc, WfC);

  // 1) project all nodes once (tables streamed exactly once)
  proj_kernel<<<ND_ / 16, 128, 0, stream>>>(drug, WfD, bd, projD, ND_, FD_);
  proj_kernel<<<NG_ / 16, 128, 0, stream>>>(gene, WfG, bgn, projG, NG_, FG_);
  proj_kernel<<<NCC_ / 16, 128, 0, stream>>>(cell, WfC, bc, projC, NCC_, FC_);

  // 2) self embeddings
  gather_rows<<<B_, 128, 0, stream>>>(projD, ids, hbuf);

  // 3) all 6 GRU candidates (h-independent)
  const float* projT[3]  = {projD, projG, projC};
  const int*   neighT[3] = {neighD, neighG, neighC};
  for (int l = 0; l < 2; ++l)
    for (int t = 0; t < 3; ++t) {
      int lt = l * 3 + t;
      gru_cand_kernel<<<B_ / 16, 128, GRU_LDS_BYTES, stream>>>(
          projT[t], neighT[t],
          gruW + (size_t)lt * D * 3 * D,
          gruU + (size_t)lt * D * 3 * D,
          grub + (size_t)lt * 3 * D,
          cand + (size_t)lt * B_ * D);
    }

  // 4) two attention layers
  attn_kernel<<<B_ / 4, 128, 0, stream>>>(hbuf, cand, att, hbuf);
  attn_kernel<<<B_ / 4, 128, 0, stream>>>(hbuf, cand + (size_t)3 * B_ * D, att + 2 * D, out);
}
